// RNNDecoder_17635135717887
// MI455X (gfx1250) — compile-verified, matrix operand order checked by reading, not run on hardware
//
#include <hip/hip_runtime.h>
#include <math.h>

typedef _Float16 h16;
typedef __attribute__((ext_vector_type(16))) _Float16 v16h;
typedef __attribute__((ext_vector_type(8)))  _Float16 v8h;
typedef __attribute__((ext_vector_type(8)))  float    v8f;

#define BD 256   // threads per block (8 wave32)
#define NB 128   // persistent grid blocks

namespace {
constexpr int Bb = 64, Tt = 512, ENC = 512, Hh = 512, Dd = 80, RFc = 2;
constexpr int CCH = 32, KW = 31;          // location conv
}

// ---------------- args ----------------
struct Args {
  // raw f32 inputs
  const float *enc, *pn_w1, *pn_b1, *pn_w2, *pn_b2;
  const float *conv_w, *v_w, *v_b;
  const float *bih0, *bhh0, *bih1, *bhh1, *out_b;
  const int   *data_len;
  // f16 converted weights (ws)
  const h16 *wdec, *wloc, *wih0, *whh0, *wih1, *whh1, *wout;
  const h16 *pm;                          // tile-swizzled (B*32*32 tiles of 256 f16)
  // state (zeroed each call)
  float *h0, *h1, *prev_out, *att_sum;
  h16   *h0h, *xcat0, *xcat1;             // xcat1 = [h1 f16 | att_c f16]
  // scratch
  float *dec, *gh0, *gh1, *gi0, *gi1, *erg, *att_w, *pnmid;
  h16   *f_h;                             // (B,T,32) conv features f16
  // outputs
  float *out1, *out2;
  // barrier
  unsigned *bar;
};

// ---------------- WMMA helpers ----------------
__device__ inline v8f zero8() {
  v8f z;
#pragma unroll
  for (int i = 0; i < 8; ++i) z[i] = 0.f;
  return z;
}

// Load a 16x32 (rows x k) f16 tile in the wave32 A/B operand layout.
__device__ inline v16h ld_tile(const h16* p, int ld) {
  const int lane = threadIdx.x & 31;
  const h16* r = p + (lane & 15) * ld + ((lane >> 4) << 3);
  v8h a = *(const v8h*)(r);
  v8h b = *(const v8h*)(r + 16);
  v16h o;
#pragma unroll
  for (int i = 0; i < 8; ++i) { o[i] = a[i]; o[i + 8] = b[i]; }
  return o;
}

// D(16x16) = X[m0:m0+16, :K] @ W[n0:n0+16, :K]^T  (both row-major f16)
__device__ inline v8f wtile(const h16* X, int ldx, const h16* W, int ldw,
                            int m0, int n0, int K) {
  v8f c = zero8();
  for (int k = 0; k < K; k += 32) {
    v16h a = ld_tile(X + m0 * ldx + k, ldx);
    v16h b = ld_tile(W + n0 * ldw + k, ldw);
    c = __builtin_amdgcn_wmma_f32_16x16x32_f16(false, a, false, b,
                                               (short)0, c, false, false);
  }
  return c;
}

// store D tile: lane holds col n=lane&15, rows m = r + 8*(lane>>4)
__device__ inline void sttile(float* Y, int ldy, int m0, int n0, v8f c,
                              const float* bias) {
  const int lane = threadIdx.x & 31, n = lane & 15, hf = lane >> 4;
  const float bv = bias ? bias[n0 + n] : 0.f;
#pragma unroll
  for (int r = 0; r < 8; ++r)
    Y[(size_t)(m0 + r + hf * 8) * ldy + n0 + n] = c[r] + bv;
}

// ---------------- device-wide barrier ----------------
__device__ inline void grid_barrier(unsigned* cnt, unsigned* gen) {
  __syncthreads();
  if (threadIdx.x == 0) {
    __threadfence();
    unsigned g = __hip_atomic_load(gen, __ATOMIC_ACQUIRE, __HIP_MEMORY_SCOPE_AGENT);
    unsigned a = __hip_atomic_fetch_add(cnt, 1u, __ATOMIC_ACQ_REL, __HIP_MEMORY_SCOPE_AGENT);
    if (a == gridDim.x - 1) {
      __hip_atomic_store(cnt, 0u, __ATOMIC_RELAXED, __HIP_MEMORY_SCOPE_AGENT);
      __hip_atomic_fetch_add(gen, 1u, __ATOMIC_ACQ_REL, __HIP_MEMORY_SCOPE_AGENT);
    } else {
      while (__hip_atomic_load(gen, __ATOMIC_ACQUIRE, __HIP_MEMORY_SCOPE_AGENT) == g)
        __builtin_amdgcn_s_sleep(2);
    }
  }
  __syncthreads();
}

__device__ inline float sigm(float x) { return 1.f / (1.f + expf(-x)); }

// ---------------- small utility kernels ----------------
__global__ void k_zero(float* p, int n) {
  for (int i = blockIdx.x * blockDim.x + threadIdx.x; i < n;
       i += gridDim.x * blockDim.x)
    p[i] = 0.f;
}

__global__ void k_cvt(const float* s, h16* d, int n) {
  for (int i = blockIdx.x * blockDim.x + threadIdx.x; i < n;
       i += gridDim.x * blockDim.x)
    d[i] = (h16)s[i];
}

// pm = enc @ att_enc_w^T + b, stored TILE-SWIZZLED:
// tile j (= rowtile*32 + coltile) occupies pm[j*256 + lane*8 + r],
// exactly the WMMA D-register order -> consumer does 1 coalesced B128/lane.
__global__ void k_pm(const h16* ench, const h16* wench, const float* encb,
                     h16* pm) {
  const int wave = (blockIdx.x * blockDim.x + threadIdx.x) >> 5;
  const int nw = (gridDim.x * blockDim.x) >> 5;
  const int lane = threadIdx.x & 31, n = lane & 15;
  const int njobs = (Bb * Tt / 16) * (Hh / 16);   // 65536
  for (int j = wave; j < njobs; j += nw) {
    const int mt = j >> 5, nt = j & 31;
    v8f c = wtile(ench, ENC, wench, ENC, mt * 16, nt * 16, ENC);
    const float bv = encb[nt * 16 + n];
    v8h pv;
#pragma unroll
    for (int r = 0; r < 8; ++r) pv[r] = (h16)(c[r] + bv);
    *(v8h*)(pm + (size_t)j * 256 + lane * 8) = pv;
  }
}

// ---------------- persistent decoder ----------------
__global__ void __launch_bounds__(BD) k_decode(Args A) {
  __shared__ float s_wconv[CCH * KW];
  __shared__ float sred[BD];
  __shared__ __align__(16) h16 s_wloc[Hh * CCH];   // 32KB: energy B operand
  __shared__ float s_vw[Hh];                       // 2KB

  const int gtid = blockIdx.x * BD + threadIdx.x;
  const int NT = gridDim.x * BD;
  const int wave = gtid >> 5;
  const int nwaves = NT >> 5;
  const int lane = threadIdx.x & 31, ln = lane & 15, hf = lane >> 4;

  unsigned* cnt = A.bar;
  unsigned* gen = A.bar + 1;

  // one-time LDS staging (step-invariant operands)
  for (int i = threadIdx.x; i < CCH * KW; i += BD) s_wconv[i] = A.conv_w[i];
  for (int i = threadIdx.x; i < (Hh * CCH) / 8; i += BD)
    ((uint4*)s_wloc)[i] = ((const uint4*)A.wloc)[i];
  for (int i = threadIdx.x; i < Hh; i += BD) s_vw[i] = A.v_w[i];
  __syncthreads();

  for (int s = 0; s < Tt; ++s) {
    // ---------- Phase A: conv features, prenet L1, dec / gh0 / gh1 GEMMs ----
    for (int p0 = gtid; p0 < Bb * Tt; p0 += NT) {
      const int b = p0 >> 9, t = p0 & 511;
      const int lenr = A.data_len[b] >> 1;
      float acc[CCH];
#pragma unroll
      for (int c = 0; c < CCH; ++c) acc[c] = 0.f;
      for (int k = 0; k < KW; ++k) {
        const int tt = t + k - (KW / 2);
        if ((unsigned)tt < (unsigned)Tt) {
          float pw = (s == 0) ? ((tt < lenr) ? 1.f / (float)lenr : 0.f)
                              : A.att_sum[b * Tt + tt];
#pragma unroll
          for (int c = 0; c < CCH; ++c) acc[c] += pw * s_wconv[c * KW + k];
        }
      }
      h16* fp = A.f_h + (size_t)p0 * CCH;
#pragma unroll
      for (int i = 0; i < 4; ++i) {
        v8h v;
#pragma unroll
        for (int j = 0; j < 8; ++j) v[j] = (h16)acc[i * 8 + j];
        *(v8h*)(fp + i * 8) = v;
      }
    }
    for (int i = gtid; i < Bb * 256; i += NT) {        // prenet layer 1
      const int b = i >> 8, o = i & 255;
      float a = A.pn_b1[o];
      const float* x = A.prev_out + b * (Dd * RFc);
      const float* w = A.pn_w1 + o * (Dd * RFc);
      for (int k = 0; k < Dd * RFc; ++k) a += x[k] * w[k];
      A.pnmid[i] = fmaxf(a, 0.f);
    }
    for (int j = wave; j < 896; j += nwaves) {         // dec | gh0 | gh1
      if (j < 128) {
        const int mt = j >> 5, nt = j & 31;
        v8f c = wtile(A.h0h, Hh, A.wdec, Hh, mt * 16, nt * 16, Hh);
        sttile(A.dec, Hh, mt * 16, nt * 16, c, nullptr);
      } else if (j < 512) {
        const int q = j - 128, mt = q / 96, nt = q % 96;
        v8f c = wtile(A.h0h, Hh, A.whh0, Hh, mt * 16, nt * 16, Hh);
        sttile(A.gh0, 3 * Hh, mt * 16, nt * 16, c, A.bhh0);
      } else {
        const int q = j - 512, mt = q / 96, nt = q % 96;
        v8f c = wtile(A.xcat1, 2 * Hh, A.whh1, Hh, mt * 16, nt * 16, Hh);
        sttile(A.gh1, 3 * Hh, mt * 16, nt * 16, c, A.bhh1);
      }
    }
    grid_barrier(cnt, gen);

    // ---------- Phase B: energy (LDS B operand, swizzled pm) + prenet L2 ----
    for (int j = wave; j < Bb * 32; j += nwaves) {
      const int b = j >> 5, tt = j & 31, t0 = tt << 4;
      v16h a = ld_tile(A.f_h + (size_t)(b * Tt + t0) * CCH, CCH);
      float accr[8];
#pragma unroll
      for (int r = 0; r < 8; ++r) accr[r] = 0.f;
      const size_t tbase = (size_t)(b * 32 + tt) * 32;   // pm tile row base
      for (int hc = 0; hc < Hh; hc += 16) {
        // B operand from LDS (ds_load_b128 x2)
        const h16* wl = &s_wloc[(hc + ln) * CCH + (hf << 3)];
        v8h b0 = *(const v8h*)(wl);
        v8h b1 = *(const v8h*)(wl + 16);
        v16h bv;
#pragma unroll
        for (int i = 0; i < 8; ++i) { bv[i] = b0[i]; bv[i + 8] = b1[i]; }
        v8f c = __builtin_amdgcn_wmma_f32_16x16x32_f16(false, a, false, bv,
                                                       (short)0, zero8(),
                                                       false, false);
        const float dv = A.dec[b * Hh + hc + ln];
        const float vw = s_vw[hc + ln];
        // pm tile: one coalesced 16B load per lane
        v8h pv = *(const v8h*)(A.pm + (tbase + (hc >> 4)) * 256 + lane * 8);
#pragma unroll
        for (int r = 0; r < 8; ++r)
          accr[r] += vw * tanhf(c[r] + (float)pv[r] + dv);
      }
#pragma unroll
      for (int r = 0; r < 8; ++r) {                    // 16-lane reduce
        float v = accr[r];
        v += __shfl_xor(v, 1);
        v += __shfl_xor(v, 2);
        v += __shfl_xor(v, 4);
        v += __shfl_xor(v, 8);
        if (ln == 0) A.erg[b * Tt + t0 + hf * 8 + r] = v + A.v_b[0];
      }
    }
    for (int i = gtid; i < Bb * Hh; i += NT) {         // prenet layer 2
      const int b = i >> 9, o = i & 511;
      float a = A.pn_b2[o];
      const float* x = A.pnmid + b * 256;
      const float* w = A.pn_w2 + o * 256;
      for (int k = 0; k < 256; ++k) a += x[k] * w[k];
      A.xcat0[b * (2 * Hh) + o] = (h16)fmaxf(a, 0.f);
    }
    grid_barrier(cnt, gen);

    // ---------- Phase C: softmax + att_sum + att output + context ----------
    if (blockIdx.x < Bb) {
      const int b = blockIdx.x;
      const int lenr = A.data_len[b] >> 1;
      float m = -3.0e38f;
      for (int t = threadIdx.x; t < Tt; t += BD)
        if (t < lenr) m = fmaxf(m, A.erg[b * Tt + t]);
      sred[threadIdx.x] = m;
      __syncthreads();
      for (int o = BD >> 1; o > 0; o >>= 1) {
        if (threadIdx.x < o)
          sred[threadIdx.x] = fmaxf(sred[threadIdx.x], sred[threadIdx.x + o]);
        __syncthreads();
      }
      m = sred[0];
      __syncthreads();
      float ssum = 0.f;
      for (int t = threadIdx.x; t < Tt; t += BD) {
        float w = (t < lenr) ? expf(A.erg[b * Tt + t] - m) : 0.f;
        A.att_w[b * Tt + t] = w;
        ssum += w;
      }
      sred[threadIdx.x] = ssum;
      __syncthreads();
      for (int o = BD >> 1; o > 0; o >>= 1) {
        if (threadIdx.x < o) sred[threadIdx.x] += sred[threadIdx.x + o];
        __syncthreads();
      }
      const float inv = 1.f / sred[0];
      __syncthreads();
      for (int t = threadIdx.x; t < Tt; t += BD) {
        float w = A.att_w[b * Tt + t] * inv;
        A.att_w[b * Tt + t] = w;
        A.att_sum[b * Tt + t] += w;
        A.out2[(size_t)(b * Tt + t) * Tt + s] = w;     // (B,T,steps)
      }
      __syncthreads();
      // context att_c[b,:] = att_w[b,:] @ enc[b,:,:]  (same block owns b)
      for (int c = threadIdx.x; c < ENC; c += BD) {
        const float* w = A.att_w + b * Tt;
        const float* e = A.enc + (size_t)b * Tt * ENC + c;
        float a = 0.f;
        for (int t = 0; t < Tt; ++t) {
          if ((t & 63) == 0) __builtin_prefetch(e + (size_t)(t + 64) * ENC, 0, 0);
          a += w[t] * e[(size_t)t * ENC];
        }
        const h16 hv = (h16)a;
        A.xcat0[b * (2 * Hh) + Hh + c] = hv;
        A.xcat1[b * (2 * Hh) + Hh + c] = hv;
      }
    }
    grid_barrier(cnt, gen);

    // ---------- Phase E: gi0 = [prenet,att_c] @ Wih0^T ----------------------
    for (int j = wave; j < 384; j += nwaves) {
      const int mt = j / 96, nt = j % 96;
      v8f c = wtile(A.xcat0, 2 * Hh, A.wih0, 2 * Hh, mt * 16, nt * 16, 2 * Hh);
      sttile(A.gi0, 3 * Hh, mt * 16, nt * 16, c, A.bih0);
    }
    grid_barrier(cnt, gen);

    // ---------- Phase F: GRU0 gate combine ----------------------------------
    for (int i = gtid; i < Bb * Hh; i += NT) {
      const int b = i >> 9, h = i & 511;
      const int gb = b * 3 * Hh;
      const float r = sigm(A.gi0[gb + h] + A.gh0[gb + h]);
      const float z = sigm(A.gi0[gb + Hh + h] + A.gh0[gb + Hh + h]);
      const float n = tanhf(A.gi0[gb + 2 * Hh + h] + r * A.gh0[gb + 2 * Hh + h]);
      const float hn = (1.f - z) * n + z * A.h0[i];
      A.h0[i] = hn;
      A.h0h[i] = (h16)hn;
    }
    grid_barrier(cnt, gen);

    // ---------- Phase G: gi1 = h0n @ Wih1^T ---------------------------------
    for (int j = wave; j < 384; j += nwaves) {
      const int mt = j / 96, nt = j % 96;
      v8f c = wtile(A.h0h, Hh, A.wih1, Hh, mt * 16, nt * 16, Hh);
      sttile(A.gi1, 3 * Hh, mt * 16, nt * 16, c, A.bih1);
    }
    grid_barrier(cnt, gen);

    // ---------- Phase H: GRU1 gate combine ----------------------------------
    for (int i = gtid; i < Bb * Hh; i += NT) {
      const int b = i >> 9, h = i & 511;
      const int gb = b * 3 * Hh;
      const float r = sigm(A.gi1[gb + h] + A.gh1[gb + h]);
      const float z = sigm(A.gi1[gb + Hh + h] + A.gh1[gb + Hh + h]);
      const float n = tanhf(A.gi1[gb + 2 * Hh + h] + r * A.gh1[gb + 2 * Hh + h]);
      const float hn = (1.f - z) * n + z * A.h1[i];
      A.h1[i] = hn;
      A.xcat1[b * (2 * Hh) + h] = (h16)hn;
    }
    grid_barrier(cnt, gen);

    // ---------- Phase I: out = [h1n,att_c] @ out_w^T; write mel + prev_out --
    for (int j = wave; j < 40; j += nwaves) {          // 4 m-tiles x 10 n-tiles
      const int mt = j / 10, nt = j % 10;
      v8f c = wtile(A.xcat1, 2 * Hh, A.wout, 2 * Hh, mt * 16, nt * 16, 2 * Hh);
      const int col = nt * 16 + ln;
      const float bo = A.out_b[col];
      const int d = col >> 1, rr = col & 1;
#pragma unroll
      for (int r = 0; r < 8; ++r) {
        const int b = mt * 16 + hf * 8 + r;
        const float v = c[r] + bo;
        A.prev_out[b * (Dd * RFc) + col] = v;
        A.out1[(size_t)b * (Dd * Tt * RFc) + (size_t)d * (Tt * RFc) + s * RFc + rr] = v;
      }
    }
    grid_barrier(cnt, gen);
  }
}

// ---------------- host launcher ----------------
extern "C" void kernel_launch(void* const* d_in, const int* in_sizes, int n_in,
                              void* d_out, int out_size, void* d_ws, size_t ws_size,
                              hipStream_t stream) {
  (void)in_sizes; (void)n_in; (void)out_size; (void)ws_size;

  const float* enc      = (const float*)d_in[0];
  const int*   data_len = (const int*)  d_in[1];
  const float* pn_w1 = (const float*)d_in[2];
  const float* pn_b1 = (const float*)d_in[3];
  const float* pn_w2 = (const float*)d_in[4];
  const float* pn_b2 = (const float*)d_in[5];
  const float* att_enc_w = (const float*)d_in[6];
  const float* att_enc_b = (const float*)d_in[7];
  const float* att_dec_w = (const float*)d_in[8];
  const float* att_conv_w = (const float*)d_in[9];
  const float* att_loc_w  = (const float*)d_in[10];
  const float* att_v_w = (const float*)d_in[11];
  const float* att_v_b = (const float*)d_in[12];
  const float* g0_wih = (const float*)d_in[13];
  const float* g0_whh = (const float*)d_in[14];
  const float* g0_bih = (const float*)d_in[15];
  const float* g0_bhh = (const float*)d_in[16];
  const float* g1_wih = (const float*)d_in[17];
  const float* g1_whh = (const float*)d_in[18];
  const float* g1_bih = (const float*)d_in[19];
  const float* g1_bhh = (const float*)d_in[20];
  const float* out_w  = (const float*)d_in[21];
  const float* out_b  = (const float*)d_in[22];

  char* base = (char*)d_ws;
  size_t off = 0;
  auto alloc = [&](size_t bytes) -> void* {
    void* p = base + off;
    off = (off + bytes + 255) & ~(size_t)255;
    return p;
  };

  // zeroed state region (barrier first)
  unsigned* bar    = (unsigned*)alloc(256);
  float* h0        = (float*)alloc(Bb * Hh * 4);
  float* h1        = (float*)alloc(Bb * Hh * 4);
  float* prev_out  = (float*)alloc(Bb * Dd * RFc * 4);
  float* att_sum   = (float*)alloc(Bb * Tt * 4);
  h16*   h0h       = (h16*)  alloc(Bb * Hh * 2);
  h16*   xcat0     = (h16*)  alloc(Bb * 2 * Hh * 2);
  h16*   xcat1     = (h16*)  alloc(Bb * 2 * Hh * 2);
  const size_t zeroBytes = off;

  // scratch (computed before use each step)
  float* dec   = (float*)alloc(Bb * Hh * 4);
  float* gh0   = (float*)alloc(Bb * 3 * Hh * 4);
  float* gh1   = (float*)alloc(Bb * 3 * Hh * 4);
  float* gi0   = (float*)alloc(Bb * 3 * Hh * 4);
  float* gi1   = (float*)alloc(Bb * 3 * Hh * 4);
  float* erg   = (float*)alloc(Bb * Tt * 4);
  float* att_w = (float*)alloc(Bb * Tt * 4);
  float* pnmid = (float*)alloc(Bb * 256 * 4);
  h16*   f_h   = (h16*)  alloc((size_t)Bb * Tt * CCH * 2);

  // f16 converted weights + tensors
  h16* wdec  = (h16*)alloc((size_t)Hh * Hh * 2);
  h16* wloc  = (h16*)alloc((size_t)Hh * CCH * 2);
  h16* wih0  = (h16*)alloc((size_t)3 * Hh * 2 * Hh * 2);
  h16* whh0  = (h16*)alloc((size_t)3 * Hh * Hh * 2);
  h16* wih1  = (h16*)alloc((size_t)3 * Hh * Hh * 2);
  h16* whh1  = (h16*)alloc((size_t)3 * Hh * Hh * 2);
  h16* wout  = (h16*)alloc((size_t)Dd * RFc * 2 * Hh * 2);
  h16* wench = (h16*)alloc((size_t)Hh * ENC * 2);
  h16* ench  = (h16*)alloc((size_t)Bb * Tt * ENC * 2);
  h16* pmh   = (h16*)alloc((size_t)Bb * Tt * Hh * 2);

  // 1) zero state + barrier
  k_zero<<<256, 256, 0, stream>>>((float*)d_ws, (int)(zeroBytes / 4));

  // 2) f16 conversions
  k_cvt<<<256, 256, 0, stream>>>(att_dec_w, wdec, Hh * Hh);
  k_cvt<<<64, 256, 0, stream>>>(att_loc_w, wloc, Hh * CCH);
  k_cvt<<<512, 256, 0, stream>>>(g0_wih, wih0, 3 * Hh * 2 * Hh);
  k_cvt<<<512, 256, 0, stream>>>(g0_whh, whh0, 3 * Hh * Hh);
  k_cvt<<<512, 256, 0, stream>>>(g1_wih, wih1, 3 * Hh * Hh);
  k_cvt<<<512, 256, 0, stream>>>(g1_whh, whh1, 3 * Hh * Hh);
  k_cvt<<<256, 256, 0, stream>>>(out_w, wout, Dd * RFc * 2 * Hh);
  k_cvt<<<256, 256, 0, stream>>>(att_enc_w, wench, Hh * ENC);
  k_cvt<<<2048, 256, 0, stream>>>(enc, ench, Bb * Tt * ENC);

  // 3) pm = enc @ W_enc^T + b  (WMMA, once, tile-swizzled output)
  k_pm<<<2048, 256, 0, stream>>>(ench, wench, att_enc_b, pmh);

  // 4) persistent sequential decoder
  Args A;
  A.enc = enc; A.pn_w1 = pn_w1; A.pn_b1 = pn_b1; A.pn_w2 = pn_w2; A.pn_b2 = pn_b2;
  A.conv_w = att_conv_w; A.v_w = att_v_w; A.v_b = att_v_b;
  A.bih0 = g0_bih; A.bhh0 = g0_bhh; A.bih1 = g1_bih; A.bhh1 = g1_bhh;
  A.out_b = out_b; A.data_len = data_len;
  A.wdec = wdec; A.wloc = wloc; A.wih0 = wih0; A.whh0 = whh0;
  A.wih1 = wih1; A.whh1 = whh1; A.wout = wout; A.pm = pmh;
  A.h0 = h0; A.h1 = h1; A.prev_out = prev_out; A.att_sum = att_sum;
  A.h0h = h0h; A.xcat0 = xcat0; A.xcat1 = xcat1;
  A.dec = dec; A.gh0 = gh0; A.gh1 = gh1; A.gi0 = gi0; A.gi1 = gi1;
  A.erg = erg; A.att_w = att_w; A.pnmid = pnmid; A.f_h = f_h;
  A.out1 = (float*)d_out;
  A.out2 = (float*)d_out + (size_t)Bb * Dd * Tt * RFc;
  A.bar = bar;

  k_decode<<<NB, BD, 0, stream>>>(A);
}
